// CVM_VIGOR_30975304138888
// MI455X (gfx1250) — matrix operand order repaired and checked by reading, not compiled, not run on hardware
//
#include <hip/hip_runtime.h>

// ---------------------------------------------------------------------------
// MSDA dual-branch fusion for MI455X (gfx1250, wave32, WMMA bf16).
//
// d_in flattening assumption (insertion order, nested dicts depth-first):
//   0: Q                     (4,16384,256) f32
//   1..4:  sat_f0..sat_f3    (4,C,H,W) f32   C in {24,40,112,1280}
//   5..8:  osm_f0..osm_f3
//   9..32:  params['sat'] : per l in 0..3: proj_w, proj_b, gn_g, gn_b (16)
//           then Wv, bv, Woff, boff, Wattn, battn, Wout, bout          (8)
//   33..56: params['osm'] same layout
//   57..62: params['ln']  : g, b, g_sat, b_sat, g_osm, b_osm
//   63: batch_size (int scalar, ignored; B fixed at 4)
//
// d_out (f32): fused(4*16384*256) | alpha(1) | sat_out(...) | osm_out(...)
// ---------------------------------------------------------------------------

#define B_      4
#define NQ_     16384
#define EMBED_  256
#define HEADS_  8
#define HD_     32
#define TTOK_   21760
#define EPS_    1e-5f

typedef unsigned short u16;
typedef unsigned int   u32;

typedef __attribute__((ext_vector_type(16))) __bf16 v16bf;
typedef __attribute__((ext_vector_type(8)))  float  v8f;

static __device__ __forceinline__ float bf2f(u16 u) {
  union { u32 i; float f; } c; c.i = ((u32)u) << 16; return c.f;
}
static __device__ __forceinline__ u16 f2bf(float f) {
  union { float f; u32 u; } c; c.f = f;
  return (u16)((c.u + 0x7fffu + ((c.u >> 16) & 1u)) >> 16);  // RNE
}

// -------------------------------- conversions ------------------------------

__global__ void cvt_f32_bf16_k(const float* __restrict__ src, u16* __restrict__ dst, size_t n) {
  size_t i = (size_t)blockIdx.x * blockDim.x + threadIdx.x;
  if (i < n) dst[i] = f2bf(src[i]);
}

// src: (K,N) row-major f32  ->  dst: (N,K) row-major bf16 (for WMMA B frags)
__global__ void transpose_cvt_k(const float* __restrict__ src, u16* __restrict__ dst, int K, int N) {
  int i = blockIdx.x * blockDim.x + threadIdx.x;
  if (i >= K * N) return;
  int n = i % N, k = i / N;
  dst[(size_t)n * K + k] = f2bf(src[i]);
}

// ------------------------- projection + GroupNorm --------------------------

// y[b, tok0+hw, o] = sum_c x[b,c,hw] * w[o,c] + bias[o]    (bf16 out)
__global__ void proj_k(const float* __restrict__ x, const float* __restrict__ w,
                       const float* __restrict__ bias, u16* __restrict__ flat,
                       int C, int HW, int tok0) {
  int i = blockIdx.x * blockDim.x + threadIdx.x;          // B*HW*256
  if (i >= B_ * HW * 256) return;
  int o  = i & 255;
  int hw = (i >> 8) % HW;
  int b  = i / (HW * 256);
  const float* xp = x + (size_t)b * C * HW + hw;
  const float* wp = w + (size_t)o * C;
  float acc = bias[o];
  for (int c = 0; c < C; ++c) acc += xp[(size_t)c * HW] * wp[c];
  flat[((size_t)(b * TTOK_ + tok0 + hw) << 8) + o] = f2bf(acc);
}

// one block per (b, group): mean/rstd over 8 channels x HW tokens
__global__ void gn_stats_k(const u16* __restrict__ flat, float* __restrict__ stats,
                           int HW, int tok0) {
  __shared__ float s1[256], s2[256];
  int b = blockIdx.x >> 5, g = blockIdx.x & 31;
  int cnt = 8 * HW;
  float sum = 0.f, sq = 0.f;
  for (int i = threadIdx.x; i < cnt; i += 256) {
    int ch = g * 8 + (i & 7), t = i >> 3;
    float v = bf2f(flat[((size_t)(b * TTOK_ + tok0 + t) << 8) + ch]);
    sum += v; sq += v * v;
  }
  s1[threadIdx.x] = sum; s2[threadIdx.x] = sq; __syncthreads();
  for (int s = 128; s > 0; s >>= 1) {
    if (threadIdx.x < s) { s1[threadIdx.x] += s1[threadIdx.x + s]; s2[threadIdx.x] += s2[threadIdx.x + s]; }
    __syncthreads();
  }
  if (threadIdx.x == 0) {
    float mu = s1[0] / cnt;
    float var = s2[0] / cnt - mu * mu;
    stats[(b * 32 + g) * 2 + 0] = mu;
    stats[(b * 32 + g) * 2 + 1] = rsqrtf(var + EPS_);
  }
}

__global__ void gn_apply_k(u16* __restrict__ flat, const float* __restrict__ stats,
                           const float* __restrict__ gg, const float* __restrict__ gb,
                           int HW, int tok0) {
  int i = blockIdx.x * blockDim.x + threadIdx.x;          // B*HW*256
  if (i >= B_ * HW * 256) return;
  int ch = i & 255;
  int t  = (i >> 8) % HW;
  int b  = i / (HW * 256);
  int g  = ch >> 3;
  size_t idx = ((size_t)(b * TTOK_ + tok0 + t) << 8) + ch;
  float mu = stats[(b * 32 + g) * 2 + 0];
  float rs = stats[(b * 32 + g) * 2 + 1];
  flat[idx] = f2bf((bf2f(flat[idx]) - mu) * rs * gg[ch] + gb[ch]);
}

// ------------------------------ WMMA GEMM ----------------------------------
// C[M,N] = A[M,K](bf16, row-major) @ W[N,K](bf16, N-major == B^T) + bias[N]
// 32x32 macro-tile per wave: 2 A-frags x 2 B-frags -> 4 WMMAs per K32 step,
// i.e. one 32B fragment load per v_wmma_f32_16x16x32_bf16 (2x better than
// the naive 16x16/wave version). M,N here are always multiples of 32.

union Frag { uint4 q[2]; v16bf v; };

static __device__ __forceinline__ v8f wmma_bf16(const Frag& a, const Frag& b, v8f c) {
  return __builtin_amdgcn_wmma_f32_16x16x32_bf16(
      false, a.v, false, b.v, (short)0, c, false, false);
}

__global__ void __launch_bounds__(256)
wmma_gemm_k(const u16* __restrict__ A, const u16* __restrict__ Wt,
            const float* __restrict__ bias,
            float* __restrict__ Cf, u16* __restrict__ Cb,
            int M, int N, int K) {
  int lane = threadIdx.x & 31;
  int wave = threadIdx.x >> 5;
  int gw = blockIdx.x * 8 + wave;
  int tn2 = N >> 5;                 // 32-wide macro-tiles along N
  int tm  = gw / tn2;
  int tn  = gw % tn2;
  if (tm >= (M >> 5)) return;       // uniform per wave -> EXEC stays all-1s

  int half = lane >> 4;             // 0: lanes 0-15, 1: lanes 16-31
  int ln16 = lane & 15;
  int kbA = half << 4;              // A: K offset 0 or 16 within 32-block
  int kbB = half << 4;              // B: K offset 0 or 16 within 32-block

  const u16* arow0 = A  + (size_t)(tm * 32 + ln16) * K;
  const u16* arow1 = arow0 + (size_t)16 * K;
  const u16* bcol0 = Wt + (size_t)(tn * 32 + ln16) * K;
  const u16* bcol1 = bcol0 + (size_t)16 * K;

  v8f acc00 = {}, acc01 = {}, acc10 = {}, acc11 = {};
  for (int kt = 0; kt < K; kt += 32) {
    Frag a0, a1, b0, b1;
    a0.q[0] = *(const uint4*)(arow0 + kt + kbA);        // e0..7  -> K=kt+kbA+e
    a0.q[1] = *(const uint4*)(arow0 + kt + kbA + 8);    // e8..15
    a1.q[0] = *(const uint4*)(arow1 + kt + kbA);
    a1.q[1] = *(const uint4*)(arow1 + kt + kbA + 8);
    b0.q[0] = *(const uint4*)(bcol0 + kt + kbB);        // e0..7  -> K=kt+kbB+e
    b0.q[1] = *(const uint4*)(bcol0 + kt + kbB + 8);
    b1.q[0] = *(const uint4*)(bcol1 + kt + kbB);
    b1.q[1] = *(const uint4*)(bcol1 + kt + kbB + 8);
    __builtin_prefetch(arow0 + kt + 64, 0, 1);          // global_prefetch_b8
    __builtin_prefetch(arow1 + kt + 64, 0, 1);
    acc00 = wmma_bf16(a0, b0, acc00);
    acc01 = wmma_bf16(a0, b1, acc01);
    acc10 = wmma_bf16(a1, b0, acc10);
    acc11 = wmma_bf16(a1, b1, acc11);
  }

  int n0 = tn * 32 + ln16, n1 = n0 + 16;
  float bi0 = bias ? bias[n0] : 0.f;
  float bi1 = bias ? bias[n1] : 0.f;
#pragma unroll
  for (int r = 0; r < 8; ++r) {
    int m0 = tm * 32 + r + (half << 3);                 // C/D: VGPR r -> M=r(+8)
    int m1 = m0 + 16;
    float v00 = acc00[r] + bi0, v01 = acc01[r] + bi1;
    float v10 = acc10[r] + bi0, v11 = acc11[r] + bi1;
    if (Cf) {
      Cf[(size_t)m0 * N + n0] = v00;  Cf[(size_t)m0 * N + n1] = v01;
      Cf[(size_t)m1 * N + n0] = v10;  Cf[(size_t)m1 * N + n1] = v11;
    } else {
      Cb[(size_t)m0 * N + n0] = f2bf(v00);  Cb[(size_t)m0 * N + n1] = f2bf(v01);
      Cb[(size_t)m1 * N + n0] = f2bf(v10);  Cb[(size_t)m1 * N + n1] = f2bf(v11);
    }
  }
}

// ------------------------------- softmax-16 --------------------------------

__global__ void softmax16_k(float* __restrict__ attn, int total) {
  int i = blockIdx.x * blockDim.x + threadIdx.x;          // B*NQ*HEADS
  if (i >= total) return;
  float* p = attn + (size_t)i * 16;
  float m = p[0];
#pragma unroll
  for (int j = 1; j < 16; ++j) m = fmaxf(m, p[j]);
  float e[16], s = 0.f;
#pragma unroll
  for (int j = 0; j < 16; ++j) { e[j] = expf(p[j] - m); s += e[j]; }
  float inv = 1.f / s;
#pragma unroll
  for (int j = 0; j < 16; ++j) p[j] = e[j] * inv;
}

// --------------------------- deformable sampling ---------------------------

__global__ void msda_sample_k(const u16* __restrict__ v, const float* __restrict__ off,
                              const float* __restrict__ attn, u16* __restrict__ samp) {
  int i = blockIdx.x * blockDim.x + threadIdx.x;          // B*NQ*HEADS
  if (i >= B_ * NQ_ * HEADS_) return;
  int h = i % HEADS_;
  int q = (i / HEADS_) % NQ_;
  int b = i / (HEADS_ * NQ_);

  const int LW[4]   = {128, 64, 32, 16};
  const int LOFF[4] = {0, 16384, 20480, 21504};

  float refx = (float)(q & 127) * (1.0f / 127.0f);
  float refy = (float)(q >> 7)  * (1.0f / 127.0f);

  const float* offp  = off  + ((size_t)(b * NQ_ + q) << 8) + h * 32;   // (L,P,2)
  const float* attnp = attn + ((size_t)(b * NQ_ + q) * 128) + h * 16;  // (L,P)

  float acc[HD_];
#pragma unroll
  for (int d = 0; d < HD_; ++d) acc[d] = 0.f;

#pragma unroll
  for (int l = 0; l < 4; ++l) {
    int W = LW[l], H = LW[l];
    float invW = 1.0f / (float)W;
    const u16* vbase = v + ((size_t)(b * TTOK_ + LOFF[l]) << 8) + h * 32;
#pragma unroll
    for (int p = 0; p < 4; ++p) {
      float ox = offp[(l * 4 + p) * 2 + 0];
      float oy = offp[(l * 4 + p) * 2 + 1];
      float a  = attnp[l * 4 + p];
      float x = (refx + ox * invW) * (float)W - 0.5f;
      float y = (refy + oy * invW) * (float)H - 0.5f;
      float x0f = floorf(x), y0f = floorf(y);
      float wx = x - x0f, wy = y - y0f;
      int x0 = (int)x0f, y0 = (int)y0f;
#pragma unroll
      for (int c = 0; c < 4; ++c) {
        int dx = c & 1, dy = c >> 1;
        int xi = x0 + dx, yi = y0 + dy;
        float wgt = (dx ? wx : 1.f - wx) * (dy ? wy : 1.f - wy);
        bool valid = (xi >= 0) && (xi < W) && (yi >= 0) && (yi < H);
        float wa = a * wgt * (valid ? 1.f : 0.f);
        int xc = min(max(xi, 0), W - 1);
        int yc = min(max(yi, 0), H - 1);
        const uint4* g4 = (const uint4*)(vbase + ((size_t)(yc * W + xc) << 8));
#pragma unroll
        for (int k = 0; k < 4; ++k) {
          uint4 d4 = g4[k];
          u32 uu[4] = {d4.x, d4.y, d4.z, d4.w};
#pragma unroll
          for (int j = 0; j < 4; ++j) {
            acc[k * 8 + j * 2 + 0] += wa * bf2f((u16)(uu[j] & 0xffffu));
            acc[k * 8 + j * 2 + 1] += wa * bf2f((u16)(uu[j] >> 16));
          }
        }
      }
    }
  }
  u16* op = samp + ((size_t)(b * NQ_ + q) << 8) + h * 32;
#pragma unroll
  for (int d = 0; d < HD_; ++d) op[d] = f2bf(acc[d]);
}

// -------------------------- fusion + 3x LayerNorm --------------------------

static __device__ float block_sum(float v, float* sbuf) {
  sbuf[threadIdx.x] = v; __syncthreads();
  for (int s = 128; s > 0; s >>= 1) {
    if (threadIdx.x < s) sbuf[threadIdx.x] += sbuf[threadIdx.x + s];
    __syncthreads();
  }
  float r = sbuf[0]; __syncthreads();
  return r;
}

__global__ void __launch_bounds__(256)
fuse_ln_k(const float* __restrict__ Q, float* __restrict__ out,
          const float* g, const float* bt, const float* gs, const float* bs,
          const float* go, const float* bo) {
  __shared__ float sbuf[256];
  const size_t STRIDE = (size_t)B_ * NQ_ * EMBED_;
  int row = blockIdx.x, e = threadIdx.x;
  size_t idx = (size_t)row * 256 + e;
  float* fused = out;
  float* satp  = out + STRIDE + 1;
  float* osmp  = out + 2 * STRIDE + 1;

  float sv = satp[idx], ov = osmp[idx], qv = Q[idx];
  float f = 0.5f * sv + 0.5f * ov + qv;
  float s = sv + qv;
  float o = ov + qv;

  float fm  = block_sum(f, sbuf) * (1.f / 256.f);
  float fv  = block_sum(f * f, sbuf) * (1.f / 256.f) - fm * fm;
  float sm  = block_sum(s, sbuf) * (1.f / 256.f);
  float sva = block_sum(s * s, sbuf) * (1.f / 256.f) - sm * sm;
  float om  = block_sum(o, sbuf) * (1.f / 256.f);
  float ova = block_sum(o * o, sbuf) * (1.f / 256.f) - om * om;

  fused[idx] = (f - fm) * rsqrtf(fv + EPS_) * g[e] + bt[e];
  satp[idx]  = (s - sm) * rsqrtf(sva + EPS_) * gs[e] + bs[e];
  osmp[idx]  = (o - om) * rsqrtf(ova + EPS_) * go[e] + bo[e];
  if (row == 0 && e == 0) out[STRIDE] = 0.5f;   // alpha
}

// ------------------------------- host side ---------------------------------

static inline int cdiv(long long a, long long b) { return (int)((a + b - 1) / b); }

static void launch_gemm(const u16* A, const u16* Wt, const float* bias,
                        float* Cf, u16* Cb, int M, int N, int K, hipStream_t s) {
  long long waves = (long long)(M >> 5) * (N >> 5);   // 32x32 macro-tiles
  wmma_gemm_k<<<cdiv(waves, 8), 256, 0, s>>>(A, Wt, bias, Cf, Cb, M, N, K);
}

extern "C" void kernel_launch(void* const* d_in, const int* in_sizes, int n_in,
                              void* d_out, int out_size, void* d_ws, size_t ws_size,
                              hipStream_t stream) {
  (void)in_sizes; (void)n_in; (void)out_size; (void)ws_size;

  const float* Q = (const float*)d_in[0];
  const int PB[2]  = {9, 33};     // sat / osm param base
  const int FB[2]  = {1, 5};      // sat / osm feature base
  const int LNB    = 57;

  const int LHW[4] = {16384, 4096, 1024, 256};
  const int LT0[4] = {0, 16384, 20480, 21504};
  const int LC[4]  = {24, 40, 112, 1280};

  // --- workspace carving -------------------------------------------------
  char* ws = (char*)d_ws;
  size_t cur = 0;
  auto carve = [&](size_t bytes) -> void* {
    void* p = ws + cur;
    cur += (bytes + 255) & ~(size_t)255;
    return p;
  };
  const size_t NQE = (size_t)B_ * NQ_ * EMBED_;          // 16,777,216
  u16*  qbf   = (u16*)carve(NQE * 2);
  u16*  flat  = (u16*)carve((size_t)B_ * TTOK_ * EMBED_ * 2);
  u16*  vbf   = (u16*)carve((size_t)B_ * TTOK_ * EMBED_ * 2);
  float* offb = (float*)carve(NQE * 4);
  float* attb = (float*)carve((size_t)B_ * NQ_ * 128 * 4);
  u16*  samp  = (u16*)carve(NQE * 2);
  u16*  wv_t   = (u16*)carve(256 * 256 * 2);
  u16*  woff_t = (u16*)carve(256 * 256 * 2);
  u16*  watt_t = (u16*)carve(256 * 128 * 2);
  u16*  wout_t = (u16*)carve(256 * 256 * 2);
  float* stats = (float*)carve(B_ * 32 * 2 * 4);

  float* dout = (float*)d_out;
  const size_t STRIDE = NQE;

  // Q -> bf16 once
  cvt_f32_bf16_k<<<cdiv(NQE, 256), 256, 0, stream>>>(Q, qbf, NQE);

  for (int br = 0; br < 2; ++br) {
    auto pf = [&](int k) { return (const float*)d_in[PB[br] + k]; };

    // per-level projection + GroupNorm into flat bf16 tokens
    for (int l = 0; l < 4; ++l) {
      const float* feat = (const float*)d_in[FB[br] + l];
      int total = B_ * LHW[l] * 256;
      proj_k<<<cdiv(total, 256), 256, 0, stream>>>(
          feat, pf(4 * l + 0), pf(4 * l + 1), flat, LC[l], LHW[l], LT0[l]);
      gn_stats_k<<<B_ * 32, 256, 0, stream>>>(flat, stats, LHW[l], LT0[l]);
      gn_apply_k<<<cdiv(total, 256), 256, 0, stream>>>(
          flat, stats, pf(4 * l + 2), pf(4 * l + 3), LHW[l], LT0[l]);
    }

    // weight transpose+convert (N-major bf16 for WMMA B fragments)
    transpose_cvt_k<<<cdiv(256 * 256, 256), 256, 0, stream>>>(pf(16), wv_t, 256, 256);
    transpose_cvt_k<<<cdiv(256 * 256, 256), 256, 0, stream>>>(pf(18), woff_t, 256, 256);
    transpose_cvt_k<<<cdiv(256 * 128, 256), 256, 0, stream>>>(pf(20), watt_t, 256, 128);
    transpose_cvt_k<<<cdiv(256 * 256, 256), 256, 0, stream>>>(pf(22), wout_t, 256, 256);

    // dense GEMMs (WMMA bf16)
    launch_gemm(flat, wv_t,   pf(17), nullptr, vbf, B_ * TTOK_, 256, 256, stream); // values -> bf16
    launch_gemm(qbf,  woff_t, pf(19), offb, nullptr, B_ * NQ_, 256, 256, stream);  // offsets
    launch_gemm(qbf,  watt_t, pf(21), attb, nullptr, B_ * NQ_, 128, 256, stream);  // attn logits

    int tot_h = B_ * NQ_ * HEADS_;
    softmax16_k<<<cdiv(tot_h, 256), 256, 0, stream>>>(attb, tot_h);
    msda_sample_k<<<cdiv(tot_h, 256), 256, 0, stream>>>(vbf, offb, attb, samp);

    // output projection straight into the branch's d_out slot (raw, pre-LN)
    float* outraw = dout + STRIDE + 1 + (size_t)br * STRIDE;
    launch_gemm(samp, wout_t, pf(23), outraw, nullptr, B_ * NQ_, 256, 256, stream);
  }

  // fusion + 3 LayerNorms + alpha
  fuse_ln_k<<<B_ * NQ_, 256, 0, stream>>>(
      Q, dout,
      (const float*)d_in[LNB + 0], (const float*)d_in[LNB + 1],
      (const float*)d_in[LNB + 2], (const float*)d_in[LNB + 3],
      (const float*)d_in[LNB + 4], (const float*)d_in[LNB + 5]);
}
